// LSTM_61821759259038
// MI455X (gfx1250) — matrix operand order repaired and checked by reading, not compiled, not run on hardware
//
#include <hip/hip_runtime.h>

// ---------------------------------------------------------------------------
// LSTM (B=4096, T=512, H=64, IN=1) + final FC, mapped onto CDNA5 WMMA.
//
// Per workgroup: 16 batch rows, 4 wave32s. Wave w computes gate columns
// n = g*64 + 16w + (lane%16) for g in {i,f,g,o}, so all four gate values for
// a given (batch, hidden) element land in the same lane & accumulator slot.
// W_hh^T lives in VGPRs as loop-invariant B fragments; h is exchanged per
// timestep through a double-buffered f16 LDS tile laid out for direct
// 16-bit A-fragment loads (ds_load_b128).
// ---------------------------------------------------------------------------

typedef _Float16 v16h __attribute__((ext_vector_type(16)));
typedef _Float16 v8h  __attribute__((ext_vector_type(8)));
typedef float    v8f  __attribute__((ext_vector_type(8)));

#define HSZ      64
#define TSTEPS   512
#define BTILE    16
#define NWAVE    4
#define NTHREADS 128

__device__ __forceinline__ float fast_sigmoid(float x) {
    float e = __expf(-x);
    return __builtin_amdgcn_rcpf(1.0f + e);
}
__device__ __forceinline__ float fast_tanh(float x) {
    float e = __expf(-2.0f * x);
    return 2.0f * __builtin_amdgcn_rcpf(1.0f + e) - 1.0f;
}

__global__ __launch_bounds__(NTHREADS) void lstm_wmma_kernel(
    const float* __restrict__ x,     // [B, T]  (IN==1)
    const float* __restrict__ W_ih,  // [256, 1]
    const float* __restrict__ W_hh,  // [256, 64] row-major
    const float* __restrict__ b_ih,  // [256]
    const float* __restrict__ b_hh,  // [256]
    const float* __restrict__ W_fc,  // [1, 64]
    const float* __restrict__ b_fc,  // [1]
    float* __restrict__ out)         // [B]
{
    // LDS: W_hh as f16 [n=256][k=64]; x tile [m=16][t=512];
    // h double buffer [2][m=16][k=64] f16; FC partials [wave][m].
    __shared__ __align__(32) _Float16 Wlds[256 * HSZ];
    __shared__ __align__(16) float    xlds[BTILE * TSTEPS];
    __shared__ __align__(16) _Float16 hbuf[2][BTILE * HSZ];
    __shared__ float fcbuf[NWAVE][BTILE];

    const int tid       = threadIdx.x;
    const int wv        = tid >> 5;          // wave id 0..3
    const int lane      = tid & 31;
    const int half      = lane >> 4;         // 0 or 1 (A/B K sub-range select)
    const int lan16     = lane & 15;
    const int tile_base = blockIdx.x * BTILE;

    // ---- one-time staging --------------------------------------------------
    for (int i = tid; i < 256 * HSZ; i += NTHREADS)
        Wlds[i] = (_Float16)W_hh[i];                      // B[k][n] == W_hh[n][k]

    for (int i = tid; i < BTILE * TSTEPS; i += NTHREADS) {
        int m = i >> 9;                                   // i / 512
        int t = i & (TSTEPS - 1);
        xlds[i] = x[(size_t)(tile_base + m) * TSTEPS + t]; // coalesced along t
    }

    for (int i = tid; i < 2 * BTILE * HSZ; i += NTHREADS)
        ((_Float16*)hbuf)[i] = (_Float16)0.0f;            // h0 = 0

    __syncthreads();

    // ---- per-lane loop-invariants -----------------------------------------
    const int jn = wv * 16 + lan16;          // hidden column this lane owns
    float bias[4], wih[4];
    v16h  Bfrag[4][2];                       // [gate][K-chunk], resident in VGPRs
#pragma unroll
    for (int g = 0; g < 4; ++g) {
        const int n = g * 64 + jn;           // gate row of W_hh
        bias[g] = b_ih[n] + b_hh[n];
        wih[g]  = W_ih[n];
#pragma unroll
        for (int kc = 0; kc < 2; ++kc) {
            // B-frag VGPR v <- K = kc*32 + half*16 + 2v (+1): contiguous 16 f16
            Bfrag[g][kc] = *(const v16h*)&Wlds[n * HSZ + kc * 32 + half * 16];
        }
    }
    const float wfc = W_fc[jn];

    float c[8], h[8];
#pragma unroll
    for (int r = 0; r < 8; ++r) { c[r] = 0.0f; h[r] = 0.0f; }

    // ---- sequential T loop -------------------------------------------------
    for (int t = 0; t < TSTEPS; ++t) {
        const int rbuf = t & 1;
        const int wbuf = rbuf ^ 1;

        // A fragments (full previous h, shared across waves via LDS).
        // 16-bit A layout: lane -> M = lan16; VGPRs 0..7 hold K = half*16+0..15
        // contiguous (per 32-wide K chunk), matching the B fragments.
        const _Float16* hb = &hbuf[rbuf][0];
        const int arow = lan16 * HSZ + half * 16;
        v8h a0lo = *(const v8h*)&hb[arow +  0];
        v8h a0hi = *(const v8h*)&hb[arow +  8];
        v8h a1lo = *(const v8h*)&hb[arow + 32];
        v8h a1hi = *(const v8h*)&hb[arow + 40];
        v16h A0 = __builtin_shufflevector(a0lo, a0hi,
                    0,1,2,3,4,5,6,7,8,9,10,11,12,13,14,15);
        v16h A1 = __builtin_shufflevector(a1lo, a1hi,
                    0,1,2,3,4,5,6,7,8,9,10,11,12,13,14,15);

        // x_t for this lane's 8 batch rows (LDS broadcast reads).
        float xv[8];
#pragma unroll
        for (int r = 0; r < 8; ++r)
            xv[r] = xlds[(r + 8 * half) * TSTEPS + t];

        // C init = bias + x * W_ih, then two chained WMMAs per gate (K=64).
        v8f acc[4];
#pragma unroll
        for (int g = 0; g < 4; ++g) {
#pragma unroll
            for (int r = 0; r < 8; ++r)
                acc[g][r] = fmaf(xv[r], wih[g], bias[g]);
        }
#pragma unroll
        for (int g = 0; g < 4; ++g) {
            acc[g] = __builtin_amdgcn_wmma_f32_16x16x32_f16(
                         false, A0, false, Bfrag[g][0], (short)0, acc[g],
                         false, false);
            acc[g] = __builtin_amdgcn_wmma_f32_16x16x32_f16(
                         false, A1, false, Bfrag[g][1], (short)0, acc[g],
                         false, false);
        }

        // Elementwise LSTM cell update (all four gates live in this lane).
#pragma unroll
        for (int r = 0; r < 8; ++r) {
            const float ig = fast_sigmoid(acc[0][r]);
            const float fg = fast_sigmoid(acc[1][r]);
            const float gg = fast_tanh   (acc[2][r]);
            const float og = fast_sigmoid(acc[3][r]);
            c[r] = fg * c[r] + ig * gg;
            h[r] = og * fast_tanh(c[r]);
            // publish h (f16) for next step's A fragments: [m][k] layout
            hbuf[wbuf][(r + 8 * half) * HSZ + jn] = (_Float16)h[r];
        }

        __syncthreads();   // single barrier per step (double-buffered h)
    }

    // ---- final FC: out[b] = sum_j h_last[b,j] * W_fc[j] + b_fc ------------
#pragma unroll
    for (int r = 0; r < 8; ++r) {
        float p = h[r] * wfc;
        p += __shfl_xor(p, 1, 32);
        p += __shfl_xor(p, 2, 32);
        p += __shfl_xor(p, 4, 32);
        p += __shfl_xor(p, 8, 32);          // sum over 16 hidden cols
        if (lan16 == 0)
            fcbuf[wv][r + 8 * half] = p;
    }
    __syncthreads();
    if (tid < BTILE) {
        float s = fcbuf[0][tid] + fcbuf[1][tid] +
                  fcbuf[2][tid] + fcbuf[3][tid] + b_fc[0];
        out[tile_base + tid] = s;
    }
}

// ---------------------------------------------------------------------------
extern "C" void kernel_launch(void* const* d_in, const int* in_sizes, int n_in,
                              void* d_out, int out_size, void* d_ws, size_t ws_size,
                              hipStream_t stream) {
    const float* x    = (const float*)d_in[0];
    const float* W_ih = (const float*)d_in[1];
    const float* W_hh = (const float*)d_in[2];
    const float* b_ih = (const float*)d_in[3];
    const float* b_hh = (const float*)d_in[4];
    const float* W_fc = (const float*)d_in[5];
    const float* b_fc = (const float*)d_in[6];
    float* out = (float*)d_out;

    const int B = in_sizes[0] / TSTEPS;      // IN == 1
    dim3 grid(B / BTILE);
    dim3 block(NTHREADS);
    hipLaunchKernelGGL(lstm_wmma_kernel, grid, block, 0, stream,
                       x, W_ih, W_hh, b_ih, b_hh, W_fc, b_fc, out);
}
